// GCNBlock_68384469287503
// MI455X (gfx1250) — matrix operand order repaired and checked by reading, not compiled, not run on hardware
//
#include <hip/hip_runtime.h>
#include <hip/hip_bf16.h>

typedef __attribute__((ext_vector_type(16))) __bf16 v16bf;
typedef __attribute__((ext_vector_type(8)))  float  v8f;

#define CH 512
#define BN_EPS_F 1e-5f

// ---------------- degree / normalization ----------------
__global__ void k_deg_init(float* __restrict__ deg, int n) {
    int i = blockIdx.x * 256 + threadIdx.x;
    if (i < n) deg[i] = 1.0f;                      // self loop
}

__global__ void k_deg_count(const int* __restrict__ ei, int E, float* __restrict__ deg) {
    int e = blockIdx.x * 256 + threadIdx.x;
    if (e < E) unsafeAtomicAdd(&deg[ei[E + e]], 1.0f);   // dst row
}

__global__ void k_dis(float* __restrict__ deg, int n) {
    int i = blockIdx.x * 256 + threadIdx.x;
    if (i < n) deg[i] = rsqrtf(deg[i]);            // deg^{-1/2}
}

// ---------------- W -> Wt: transpose + convert to bf16 ----------------
// Wt[n*512 + k] = bf16(W[k*512 + n]);  B fragments become two contiguous 16B loads.
__global__ void k_wcvt(const float* __restrict__ W, __bf16* __restrict__ Wt) {
    int idx = blockIdx.x * 256 + threadIdx.x;      // over 512*512
    int n = idx >> 9;
    int k = idx & (CH - 1);
    Wt[idx] = (__bf16)W[(size_t)k * CH + n];
}

// ---------------- GEMM: h = x @ W (bf16 WMMA, f32 accum) ----------------
// grid = (CH/16, ceil(mTiles/8)), block = 256 (8 waves); one 16x16 tile / wave
__global__ void __launch_bounds__(256)
k_gemm_wmma(const float* __restrict__ x, const __bf16* __restrict__ Wt,
            float* __restrict__ h, int mTiles, int N) {
    const int lane = threadIdx.x & 31;
    const int wave = threadIdx.x >> 5;
    const int mTile = blockIdx.y * 8 + wave;
    if (mTile >= mTiles) return;                   // wave-uniform: EXEC stays full
    const int nTile = blockIdx.x;

    const int l16 = lane & 15;
    const int hl  = lane >> 4;
    int rowA = mTile * 16 + l16;
    if (rowA > N - 1) rowA = N - 1;                // clamp, keep EXEC all-ones
    const int col = nTile * 16 + l16;

    const float* __restrict__ xrow = x + (size_t)rowA * CH;
    // per-lane B column slice: K values hl*8..hl*8+7 and 16+hl*8..16+hl*8+7
    const __bf16* __restrict__ wcol = Wt + (size_t)col * CH + hl * 8;

    v8f acc = {};
    #pragma unroll 4
    for (int kb = 0; kb < CH; kb += 32) {
        // ---- A fragment (16x32 bf16), hardware f32->bf16 converts ----
        const int g0 = kb + hl * 8;
        float4 a0 = *(const float4*)(xrow + g0);
        float4 a1 = *(const float4*)(xrow + g0 + 4);
        float4 a2 = *(const float4*)(xrow + g0 + 16);
        float4 a3 = *(const float4*)(xrow + g0 + 20);
        v16bf a;
        a[0]  = (__bf16)a0.x; a[1]  = (__bf16)a0.y; a[2]  = (__bf16)a0.z; a[3]  = (__bf16)a0.w;
        a[4]  = (__bf16)a1.x; a[5]  = (__bf16)a1.y; a[6]  = (__bf16)a1.z; a[7]  = (__bf16)a1.w;
        a[8]  = (__bf16)a2.x; a[9]  = (__bf16)a2.y; a[10] = (__bf16)a2.z; a[11] = (__bf16)a2.w;
        a[12] = (__bf16)a3.x; a[13] = (__bf16)a3.y; a[14] = (__bf16)a3.z; a[15] = (__bf16)a3.w;

        // ---- B fragment (32x16 bf16): two 16B loads ----
        typedef __attribute__((ext_vector_type(8))) __bf16 v8bf_;
        v8bf_ b0 = *(const v8bf_*)(wcol + kb);
        v8bf_ b1 = *(const v8bf_*)(wcol + kb + 16);
        v16bf b = __builtin_shufflevector(b0, b1, 0,1,2,3,4,5,6,7,8,9,10,11,12,13,14,15);

        acc = __builtin_amdgcn_wmma_f32_16x16x32_bf16(
                  false, a, false, b, (short)0, acc, false, false);
    }

    // ---- store C: VGPR r -> row mTile*16 + hl*8 + r, col 'col' ----
    const int rbase = mTile * 16 + hl * 8;
    float* __restrict__ hp = h + (size_t)rbase * CH + col;
    if (mTile * 16 + 16 <= N) {                    // uniform: full tile, no guards
        #pragma unroll
        for (int r = 0; r < 8; ++r) hp[(size_t)r * CH] = acc[r];
    } else {
        #pragma unroll
        for (int r = 0; r < 8; ++r) {
            if (rbase + r < N) hp[(size_t)r * CH] = acc[r];
        }
    }
}

// ---------------- out = b + h * dis^2 (bias + self-loop), float4 ----------------
__global__ void k_out_init(const float* __restrict__ h, const float* __restrict__ dis,
                           const float* __restrict__ bias, float* __restrict__ out,
                           int total4) {
    int idx = blockIdx.x * 256 + threadIdx.x;      // over total/4
    if (idx < total4) {
        int i  = idx >> 7;                         // row (512/4 = 128 float4 per row)
        int c4 = (idx & 127) << 2;
        float d = dis[i];
        float d2 = d * d;
        float4 hv = *(const float4*)(h + ((size_t)idx << 2));
        float4 bv = *(const float4*)(bias + c4);
        float4 o;
        o.x = bv.x + hv.x * d2; o.y = bv.y + hv.y * d2;
        o.z = bv.z + hv.z * d2; o.w = bv.w + hv.w * d2;
        *(float4*)(out + ((size_t)idx << 2)) = o;
    }
}

// ---------------- edge scatter: out[dst] += dis[src]*dis[dst] * h[src] ----------------
__global__ void __launch_bounds__(128)
k_scatter(const int* __restrict__ ei, int E,
          const float* __restrict__ h, const float* __restrict__ dis,
          float* __restrict__ out) {
    const int e   = blockIdx.x;
    const int src = ei[e];
    const int dst = ei[E + e];
    const float norm = dis[src] * dis[dst];
    const int c = threadIdx.x << 2;                // 0..508 step 4
    float4 v = *(const float4*)(h + (size_t)src * CH + c);
    float* o = out + (size_t)dst * CH + c;
    unsafeAtomicAdd(o + 0, v.x * norm);
    unsafeAtomicAdd(o + 1, v.y * norm);
    unsafeAtomicAdd(o + 2, v.z * norm);
    unsafeAtomicAdd(o + 3, v.w * norm);
}

// ---------------- BatchNorm (eval) + ReLU, in place, float4 ----------------
__global__ void k_bn_relu(float* __restrict__ out,
                          const float* __restrict__ gamma, const float* __restrict__ beta,
                          const float* __restrict__ rm, const float* __restrict__ rv,
                          int total4) {
    int idx = blockIdx.x * 256 + threadIdx.x;      // over total/4
    if (idx < total4) {
        int c4 = (idx & 127) << 2;
        float4 g = *(const float4*)(gamma + c4);
        float4 bt = *(const float4*)(beta + c4);
        float4 m = *(const float4*)(rm + c4);
        float4 vv = *(const float4*)(rv + c4);
        float4 v = *(float4*)(out + ((size_t)idx << 2));
        float4 r;
        r.x = (v.x - m.x) * (g.x * rsqrtf(vv.x + BN_EPS_F)) + bt.x;
        r.y = (v.y - m.y) * (g.y * rsqrtf(vv.y + BN_EPS_F)) + bt.y;
        r.z = (v.z - m.z) * (g.z * rsqrtf(vv.z + BN_EPS_F)) + bt.z;
        r.w = (v.w - m.w) * (g.w * rsqrtf(vv.w + BN_EPS_F)) + bt.w;
        r.x = r.x > 0.f ? r.x : 0.f; r.y = r.y > 0.f ? r.y : 0.f;
        r.z = r.z > 0.f ? r.z : 0.f; r.w = r.w > 0.f ? r.w : 0.f;
        *(float4*)(out + ((size_t)idx << 2)) = r;
    }
}

extern "C" void kernel_launch(void* const* d_in, const int* in_sizes, int n_in,
                              void* d_out, int out_size, void* d_ws, size_t ws_size,
                              hipStream_t stream) {
    const float* x     = (const float*)d_in[0];
    const int*   ei    = (const int*)d_in[1];     // [2, E] (src row, dst row)
    const float* W     = (const float*)d_in[2];
    const float* bias  = (const float*)d_in[3];
    const float* gamma = (const float*)d_in[4];
    const float* beta  = (const float*)d_in[5];
    const float* rm    = (const float*)d_in[6];
    const float* rv    = (const float*)d_in[7];

    const int N = in_sizes[0] / CH;
    const int E = in_sizes[1] / 2;
    const int total = N * CH;
    const int total4 = total / 4;

    float* out = (float*)d_out;
    float* h   = (float*)d_ws;                    // [N, 512] f32
    float* dis = h + (size_t)N * CH;              // [N] f32 (deg, then deg^-1/2)
    // Wt after dis, rounded up to 32B alignment (in floats)
    size_t wtOffF = (((size_t)N * CH + N) + 7) & ~(size_t)7;
    __bf16* Wt = (__bf16*)((float*)d_ws + wtOffF); // [512, 512] bf16, transposed

    // 1) degrees of A + I, then deg^{-1/2}
    k_deg_init <<<(N + 255) / 256, 256, 0, stream>>>(dis, N);
    k_deg_count<<<(E + 255) / 256, 256, 0, stream>>>(ei, E, dis);
    k_dis      <<<(N + 255) / 256, 256, 0, stream>>>(dis, N);

    // 2) W -> Wt (bf16, transposed), then h = x @ W via WMMA
    k_wcvt<<<(CH * CH) / 256, 256, 0, stream>>>(W, Wt);
    const int mTiles = (N + 15) / 16;
    dim3 ggrid(CH / 16, (mTiles + 7) / 8);
    k_gemm_wmma<<<ggrid, 256, 0, stream>>>(x, Wt, h, mTiles, N);

    // 3) out = b + h * dis^2 (self loop), scatter edges, BN+ReLU
    k_out_init<<<(total4 + 255) / 256, 256, 0, stream>>>(h, dis, bias, out, total4);
    k_scatter <<<E, 128, 0, stream>>>(ei, E, h, dis, out);
    k_bn_relu <<<(total4 + 255) / 256, 256, 0, stream>>>(out, gamma, beta, rm, rv, total4);
}
